// MultiHeadAttention_89928025244060
// MI455X (gfx1250) — matrix operand order repaired and checked by reading, not compiled, not run on hardware
//
#include <hip/hip_runtime.h>

// ---------------------------------------------------------------------------
// MHA forward, B=4 S=2048 E=1024 H=16 Dh=64, gfx1250 (MI455X), wave32.
// fused QKV projection (12 bf16 WMMA / k-step, x staged once) ->
// causal flash attention (Br=128, online softmax in exp2 domain) ->
// out_proj (bf16 WMMA + bias). Fragment loads are contiguous 32-B runs
// (2x ds_load_b128); V stored transposed [B,H,Dh,S]; f32->bf16 via
// v_add(0x8000)+v_perm_b32 pair packing.
// ---------------------------------------------------------------------------

namespace {
constexpr int cB  = 4;
constexpr int cS  = 2048;
constexpr int cE  = 1024;
constexpr int cH  = 16;
constexpr int cDH = 64;
constexpr int cM  = cB * cS;  // 8192
}

typedef __attribute__((ext_vector_type(16))) __bf16 v16bf;
typedef __attribute__((ext_vector_type(8)))  float  v8f;

// ---- bf16 conversion helpers -------------------------------------------
__device__ __forceinline__ unsigned short f2bfu(float f) {
  return (unsigned short)((__builtin_bit_cast(unsigned int, f) + 0x8000u) >> 16);
}

#if __has_builtin(__builtin_amdgcn_cvt_pk_bf16_f32)
typedef __attribute__((ext_vector_type(2))) __bf16 v2bf;
__device__ __forceinline__ unsigned int pkbf(float lo, float hi) {
  v2bf r = __builtin_amdgcn_cvt_pk_bf16_f32(lo, hi);
  return __builtin_bit_cast(unsigned int, r);
}
#else
__device__ __forceinline__ unsigned int pkbf(float lo, float hi) {
  const unsigned int a = __builtin_bit_cast(unsigned int, lo) + 0x8000u;
  const unsigned int b = __builtin_bit_cast(unsigned int, hi) + 0x8000u;
  return __builtin_amdgcn_perm(b, a, 0x07060302u);  // {b[3:2], a[3:2]}
}
#endif

__device__ __forceinline__ v8f zero8() {
  v8f z;
#pragma unroll
  for (int i = 0; i < 8; ++i) z[i] = 0.0f;
  return z;
}

__device__ __forceinline__ v8f wmma_bf16(v16bf a, v16bf b, v8f c) {
  return __builtin_amdgcn_wmma_f32_16x16x32_bf16(false, a, false, b,
                                                 (short)0, c, false, false);
}

// ---- wide fragment loaders ---------------------------------------------
__device__ __forceinline__ v16bf loadA_w(const unsigned short* p, int ld,
                                         int lane) {
  const int half = (lane >> 4) & 1;
  const int r    = lane & 15;
  const uint4 lo = *reinterpret_cast<const uint4*>(p + r * ld + 16 * half);
  const uint4 hi = *reinterpret_cast<const uint4*>(p + r * ld + 16 * half + 8);
  v16bf a;
  __builtin_memcpy(&a, &lo, 16);
  __builtin_memcpy(reinterpret_cast<char*>(&a) + 16, &hi, 16);
  return a;
}

__device__ __forceinline__ v16bf loadB_w(const unsigned short* p, int ld,
                                         int lane) {
  const int half = (lane >> 4) & 1;
  const int c    = lane & 15;
  return *reinterpret_cast<const v16bf*>(p + c * ld + 16 * half);
}

// ---- 16-lane-group reductions ------------------------------------------
__device__ __forceinline__ float rmax16(float v) {
#pragma unroll
  for (int off = 8; off; off >>= 1) v = fmaxf(v, __shfl_xor(v, off, 32));
  return v;
}
__device__ __forceinline__ float rsum16(float v) {
#pragma unroll
  for (int off = 8; off; off >>= 1) v += __shfl_xor(v, off, 32);
  return v;
}

// ===========================================================================
// Kernel 1 (fused QKV): stage X once, 12 WMMAs per k-step.
// Qo,Ko bf16 [B,H,S,DH]; VTo bf16 [B,H,DH,S].
// Q pre-scaled by qScale = log2(e)/sqrt(Dh) (softmax done in exp2 domain).
// 256 thr / 8 waves; tile 128(M) x 64(N) x {Q,K,V}.
// ===========================================================================
__global__ __launch_bounds__(256) void qkv_fused_kernel(
    const float* __restrict__ x, const float* __restrict__ Wq,
    const float* __restrict__ Wk, const float* __restrict__ Wv,
    unsigned short* __restrict__ Qo, unsigned short* __restrict__ Ko,
    unsigned short* __restrict__ VTo, float qScale) {
  __shared__ __align__(32) unsigned short smem[10240];  // 20KB
  unsigned short* Xs = smem;  // 4096 ushort, [m][k] ld=32
  unsigned short* WtArr[3] = {smem + 4096, smem + 6144, smem + 8192};  // [n][k]

  const int tid  = threadIdx.x;
  const int lane = tid & 31;
  const int wid  = tid >> 5;
  const int wm   = wid & 3;
  const int wn   = wid >> 2;
  const int half = (lane >> 4) & 1;
  const int mBase = blockIdx.x * 128;
  const int h     = blockIdx.y;
  const float* Wh[3] = {Wq + (size_t)h * cE * cDH, Wk + (size_t)h * cE * cDH,
                        Wv + (size_t)h * cE * cDH};

  v8f acc[3][2][2];
#pragma unroll
  for (int w = 0; w < 3; ++w)
#pragma unroll
    for (int mi = 0; mi < 2; ++mi)
#pragma unroll
      for (int ni = 0; ni < 2; ++ni) acc[w][mi][ni] = zero8();

  for (int k0 = 0; k0 < cE; k0 += 32) {
    __syncthreads();
    // X tile 128x32 fp32 -> bf16 (staged once for all of Q,K,V).
#pragma unroll
    for (int i = 0; i < 4; ++i) {
      const int f   = tid + i * 256;
      const int row = f >> 3, c4 = f & 7;
      const float4 v =
          *(reinterpret_cast<const float4*>(x + (size_t)(mBase + row) * cE + k0) + c4);
      uint2 t;
      t.x = pkbf(v.x, v.y);
      t.y = pkbf(v.z, v.w);
      reinterpret_cast<uint2*>(Xs)[row * 8 + c4] = t;
    }
    // 3 weight tiles 32x64 -> column-major Wt[n][k], packed b32 stores.
    {
      const int kk = tid >> 4;  // 0..15
      const int k  = kk * 2;
      const int c4 = tid & 15;
#pragma unroll
      for (int w = 0; w < 3; ++w) {
        const float4 fa = *(reinterpret_cast<const float4*>(
                                Wh[w] + (size_t)k0 * cDH) + (k * 16 + c4));
        const float4 fb = *(reinterpret_cast<const float4*>(
                                Wh[w] + (size_t)k0 * cDH) + ((k + 1) * 16 + c4));
        const float* pa = &fa.x;
        const float* pb = &fb.x;
#pragma unroll
        for (int j = 0; j < 4; ++j) {
          const int n = c4 * 4 + j;
          reinterpret_cast<unsigned int*>(WtArr[w])[n * 16 + kk] =
              pkbf(pa[j], pb[j]);
        }
      }
    }
    if (k0 + 32 < cE)
      __builtin_prefetch(x + (size_t)(mBase + (tid >> 1)) * cE + k0 + 32, 0, 3);
    __syncthreads();

    v16bf bfr[3][2];
#pragma unroll
    for (int w = 0; w < 3; ++w)
#pragma unroll
      for (int ni = 0; ni < 2; ++ni)
        bfr[w][ni] = loadB_w(WtArr[w] + (wn * 32 + ni * 16) * 32, 32, lane);
#pragma unroll
    for (int mi = 0; mi < 2; ++mi) {
      const v16bf a = loadA_w(Xs + (wm * 32 + mi * 16) * 32, 32, lane);
#pragma unroll
      for (int w = 0; w < 3; ++w)
#pragma unroll
        for (int ni = 0; ni < 2; ++ni)
          acc[w][mi][ni] = wmma_bf16(a, bfr[w][ni], acc[w][mi][ni]);
    }
  }

  // ---- epilogue: per matrix, bounce 128x64 tile via LDS, b128 stores ----
  const int    bi    = mBase >> 11;  // / S
  const int    sBase = mBase & (cS - 1);
  const size_t bh    = (size_t)bi * cH + h;
  unsigned short* const outs[3] = {Qo, Ko, VTo};
  const float scl3[3] = {qScale, 1.0f, 1.0f};
#pragma unroll
  for (int w = 0; w < 3; ++w) {
    const bool vT = (w == 2);
    __syncthreads();
    if (vT) {
      // col-major [d][row] ld=128: lane's 8 elements consecutive -> uint4.
#pragma unroll
      for (int mi = 0; mi < 2; ++mi)
#pragma unroll
        for (int ni = 0; ni < 2; ++ni) {
          const int dl      = wn * 32 + ni * 16 + (lane & 15);
          const int rowBase = wm * 32 + mi * 16 + 8 * half;
          const v8f a = acc[w][mi][ni];
          uint4 u;
          u.x = pkbf(a[0], a[1]);
          u.y = pkbf(a[2], a[3]);
          u.z = pkbf(a[4], a[5]);
          u.w = pkbf(a[6], a[7]);
          *reinterpret_cast<uint4*>(smem + dl * 128 + rowBase) = u;
        }
    } else {
      const float s = scl3[w];
#pragma unroll
      for (int mi = 0; mi < 2; ++mi)
#pragma unroll
        for (int ni = 0; ni < 2; ++ni)
#pragma unroll
          for (int i = 0; i < 8; ++i) {
            const int rowL = wm * 32 + mi * 16 + i + 8 * half;
            const int dl   = wn * 32 + ni * 16 + (lane & 15);
            smem[rowL * 64 + dl] = f2bfu(acc[w][mi][ni][i] * s);
          }
    }
    __syncthreads();
    unsigned short* const out = outs[w];
#pragma unroll
    for (int t = 0; t < 4; ++t) {
      const int idx = tid + t * 256;  // 1024 uint4
      if (!vT) {
        const int rowL = idx >> 3, c = idx & 7;
        *reinterpret_cast<uint4*>(out + (bh * cS + sBase + rowL) * cDH + c * 8) =
            reinterpret_cast<const uint4*>(smem)[idx];
      } else {
        const int d = idx >> 4, c = idx & 15;
        *reinterpret_cast<uint4*>(out + (bh * cDH + d) * cS + sBase + c * 8) =
            *reinterpret_cast<const uint4*>(smem + d * 128 + c * 8);
      }
    }
  }
}

// ===========================================================================
// Kernel 2: causal flash attention, Br=128 (8 waves), Bc=64, exp2 domain.
// Q,K bf16 [B,H,S,DH]; VT bf16 [B,H,DH,S]; AO bf16 [B,S,H*DH].
// Grid (S/128, H, B), 256 thr; wave w owns q-rows qb*128+w*16 .. +15.
// ===========================================================================
__global__ __launch_bounds__(256) void flash_attn_kernel(
    const unsigned short* __restrict__ Q, const unsigned short* __restrict__ K,
    const unsigned short* __restrict__ VT, unsigned short* __restrict__ AO) {
  __shared__ __align__(32) unsigned short Ks[64 * 64];  // [t][d]
  __shared__ __align__(32) unsigned short Vt[64 * 64];  // [dv][t]
  __shared__ __align__(32) unsigned short Ps[8 * 16 * 64];

  const int tid  = threadIdx.x;
  const int lane = tid & 31;
  const int wid  = tid >> 5;
  const int half = (lane >> 4) & 1;
  const int qb   = blockIdx.x;
  const int h    = blockIdx.y;
  const int b    = blockIdx.z;

  const size_t headOff = ((size_t)b * cH + h) * cS * cDH;
  const unsigned short* Qh  = Q + headOff;
  const unsigned short* Kh  = K + headOff;
  const unsigned short* VTh = VT + headOff;  // [DH][S]
  const int qRow = qb * 128 + wid * 16;      // this wave's first q row

  v16bf qf[2];
  qf[0] = loadA_w(Qh + (size_t)qRow * cDH, cDH, lane);
  qf[1] = loadA_w(Qh + (size_t)qRow * cDH + 32, cDH, lane);

  float m_i[8], l_i[8];
  v8f o[4];
#pragma unroll
  for (int i = 0; i < 8; ++i) { m_i[i] = -1e30f; l_i[i] = 0.0f; }
#pragma unroll
  for (int n = 0; n < 4; ++n) o[n] = zero8();

  unsigned short* Pw = Ps + wid * 16 * 64;
  const int kbMax = 2 * qb + 1;  // key tiles 0 .. 2qb+1 cover keys <= qb*128+127

  for (int kb = 0; kb <= kbMax; ++kb) {
    __syncthreads();
    {  // stage K tile [t][d] and V tile [dv][t]; 512 uint4 each, 2/thread
      const int kBase = kb * 64;
      const uint4* Kg = reinterpret_cast<const uint4*>(Kh + (size_t)kBase * cDH);
      uint4* Ks4 = reinterpret_cast<uint4*>(Ks);
      uint4* Vt4 = reinterpret_cast<uint4*>(Vt);
#pragma unroll
      for (int i = 0; i < 2; ++i) {
        const int idx = tid + i * 256;
        Ks4[idx] = Kg[idx];
        const int dv = idx >> 3, c = idx & 7;
        Vt4[idx] =
            *(reinterpret_cast<const uint4*>(VTh + (size_t)dv * cS + kBase) + c);
      }
    }
    __syncthreads();

    // wave-uniform: does this key tile contribute to any of our 16 rows?
    const bool active = (kb * 64) <= (qRow + 15);
    if (active) {
      // ---- S = Q K^T ----
      float p[4][8];
      const bool needMask = (kb * 64 + 63) > qRow;
#pragma unroll
      for (int n = 0; n < 4; ++n) {
        v8f s = zero8();
#pragma unroll
        for (int ks = 0; ks < 2; ++ks) {
          const v16bf bfr = loadB_w(Ks + (n * 16) * 64 + ks * 32, 64, lane);
          s = wmma_bf16(qf[ks], bfr, s);
        }
        if (needMask) {
          const int tG = kb * 64 + n * 16 + (lane & 15);
#pragma unroll
          for (int i = 0; i < 8; ++i) {
            const int rG = qRow + i + 8 * half;
            if (tG > rG) s[i] = -1e30f;
          }
        }
#pragma unroll
        for (int i = 0; i < 8; ++i) p[n][i] = s[i];
      }

      // ---- online softmax (exp2 domain; log2(e) folded into Q) ----
      float mnew[8], scl[8], rs[8];
#pragma unroll
      for (int i = 0; i < 8; ++i) {
        float mt = p[0][i];
#pragma unroll
        for (int n = 1; n < 4; ++n) mt = fmaxf(mt, p[n][i]);
        mt      = rmax16(mt);
        mnew[i] = fmaxf(m_i[i], mt);
        scl[i]  = exp2f(m_i[i] - mnew[i]);
        rs[i]   = 0.0f;
      }
#pragma unroll
      for (int n = 0; n < 4; ++n)
#pragma unroll
        for (int i = 0; i < 8; ++i) {
          const float pv = exp2f(p[n][i] - mnew[i]);
          p[n][i] = pv;
          rs[i] += pv;
        }
#pragma unroll
      for (int i = 0; i < 8; ++i) {
        rs[i]  = rsum16(rs[i]);
        l_i[i] = l_i[i] * scl[i] + rs[i];
        m_i[i] = mnew[i];
      }
#pragma unroll
      for (int n = 0; n < 4; ++n)
#pragma unroll
        for (int i = 0; i < 8; ++i) o[n][i] *= scl[i];

      // ---- P -> LDS row-major for A-frag reload ----
#pragma unroll
      for (int n = 0; n < 4; ++n)
#pragma unroll
        for (int i = 0; i < 8; ++i)
          Pw[(i + 8 * half) * 64 + n * 16 + (lane & 15)] = f2bfu(p[n][i]);
    }
    __syncthreads();
    if (active) {
      // ---- O += P V ----
#pragma unroll
      for (int ks = 0; ks < 2; ++ks) {
        const v16bf a = loadA_w(Pw + ks * 32, 64, lane);
#pragma unroll
        for (int n = 0; n < 4; ++n) {
          const v16bf bfr = loadB_w(Vt + (n * 16) * 64 + ks * 32, 64, lane);
          o[n] = wmma_bf16(a, bfr, o[n]);
        }
      }
    }
  }

  // ---- epilogue: normalize, bounce via LDS, coalesced b128 stores ----
  __syncthreads();
#pragma unroll
  for (int n = 0; n < 4; ++n)
#pragma unroll
    for (int i = 0; i < 8; ++i)
      Pw[(i + 8 * half) * 64 + n * 16 + (lane & 15)] =
          f2bfu(o[n][i] * (1.0f / l_i[i]));
  __syncthreads();
#pragma unroll
  for (int j = 0; j < 4; ++j) {
    const int idx = lane + j * 32;  // 128 uint4 per wave tile
    const int row = idx >> 3, c = idx & 7;
    *reinterpret_cast<uint4*>(
        AO + ((size_t)b * cS + qRow + row) * cE + h * cDH + c * 8) =
        *reinterpret_cast<const uint4*>(Pw + row * 64 + c * 8);
  }
}

// ===========================================================================
// Kernel 3: out = AO @ Wo + bo.  AO bf16 [8192,1024]; Wo fp32 [1024,1024].
// ===========================================================================
__global__ __launch_bounds__(256) void out_proj_kernel(
    const unsigned short* __restrict__ AO, const float* __restrict__ Wo,
    const float* __restrict__ bo, float* __restrict__ out) {
  __shared__ __align__(32) float fsm[8192];  // 32KB; staging uses first 12KB
  unsigned short* As = reinterpret_cast<unsigned short*>(fsm);         // 4096
  unsigned short* Wt = reinterpret_cast<unsigned short*>(fsm) + 4096;  // 2048

  const int tid  = threadIdx.x;
  const int lane = tid & 31;
  const int wid  = tid >> 5;
  const int wm   = wid & 3;
  const int wn   = wid >> 2;
  const int half = (lane >> 4) & 1;
  const int mBase = blockIdx.x * 128;
  const int nBase = blockIdx.y * 64;

  const float bias0 = bo[nBase + wn * 32 + (lane & 15)];
  const float bias1 = bo[nBase + wn * 32 + 16 + (lane & 15)];

  v8f acc[2][2];
#pragma unroll
  for (int mi = 0; mi < 2; ++mi)
#pragma unroll
    for (int ni = 0; ni < 2; ++ni) acc[mi][ni] = zero8();

  for (int k0 = 0; k0 < cE; k0 += 32) {
    __syncthreads();
#pragma unroll
    for (int i = 0; i < 2; ++i) {
      const int idx = tid + i * 256;
      const int row = idx >> 2, c = idx & 3;
      reinterpret_cast<uint4*>(As)[idx] =
          *(reinterpret_cast<const uint4*>(AO + (size_t)(mBase + row) * cE + k0) + c);
    }
    {
      const int kk = tid >> 4;
      const int k  = kk * 2;
      const int c4 = tid & 15;
      const float4 fa = *(reinterpret_cast<const float4*>(
                              Wo + (size_t)(k0 + k) * cE + nBase) + c4);
      const float4 fb = *(reinterpret_cast<const float4*>(
                              Wo + (size_t)(k0 + k + 1) * cE + nBase) + c4);
      const float* pa = &fa.x;
      const float* pb = &fb.x;
#pragma unroll
      for (int j = 0; j < 4; ++j) {
        const int n = c4 * 4 + j;
        reinterpret_cast<unsigned int*>(Wt)[n * 16 + kk] = pkbf(pa[j], pb[j]);
      }
    }
    if (k0 + 32 < cE)
      __builtin_prefetch(AO + (size_t)(mBase + (tid >> 1)) * cE + k0 + 32, 0, 3);
    __syncthreads();

    v16bf bfr[2];
#pragma unroll
    for (int ni = 0; ni < 2; ++ni)
      bfr[ni] = loadB_w(Wt + (wn * 32 + ni * 16) * 32, 32, lane);
#pragma unroll
    for (int mi = 0; mi < 2; ++mi) {
      const v16bf a = loadA_w(As + (wm * 32 + mi * 16) * 32, 32, lane);
#pragma unroll
      for (int ni = 0; ni < 2; ++ni)
        acc[mi][ni] = wmma_bf16(a, bfr[ni], acc[mi][ni]);
    }
  }

  __syncthreads();
#pragma unroll
  for (int mi = 0; mi < 2; ++mi)
#pragma unroll
    for (int ni = 0; ni < 2; ++ni) {
      const float bias = ni ? bias1 : bias0;
#pragma unroll
      for (int i = 0; i < 8; ++i) {
        const int rowL = wm * 32 + mi * 16 + i + 8 * half;
        const int dl   = wn * 32 + ni * 16 + (lane & 15);
        fsm[rowL * 64 + dl] = acc[mi][ni][i] + bias;
      }
    }
  __syncthreads();
#pragma unroll
  for (int t = 0; t < 8; ++t) {
    const int idx = tid + t * 256;  // 2048 float4
    const int row = idx >> 4, c = idx & 15;
    *(reinterpret_cast<float4*>(out + (size_t)(mBase + row) * cE + nBase + c * 4)) =
        *(reinterpret_cast<const float4*>(fsm + row * 64 + c * 4));
  }
}

// ===========================================================================
extern "C" void kernel_launch(void* const* d_in, const int* in_sizes, int n_in,
                              void* d_out, int out_size, void* d_ws,
                              size_t ws_size, hipStream_t stream) {
  (void)in_sizes; (void)n_in; (void)out_size; (void)ws_size;
  const float* x  = (const float*)d_in[0];
  const float* Wq = (const float*)d_in[1];
  const float* Wk = (const float*)d_in[2];
  const float* Wv = (const float*)d_in[3];
  const float* Wo = (const float*)d_in[4];
  const float* bo = (const float*)d_in[5];
  float* out = (float*)d_out;

  // ws (bf16 as ushort): Q | K | V^T | attn-out   (64 MB total)
  const size_t nQKV = (size_t)cB * cH * cS * cDH;
  unsigned short* Qw  = (unsigned short*)d_ws;
  unsigned short* Kw  = Qw + nQKV;
  unsigned short* VTw = Kw + nQKV;
  unsigned short* AOw = VTw + nQKV;

  // Q scale folds 1/sqrt(Dh) and log2(e) (flash softmax runs in exp2 domain)
  const float qScale = 0.125f * 1.44269504088896340736f;

  qkv_fused_kernel<<<dim3(cM / 128, cH), 256, 0, stream>>>(
      x, Wq, Wk, Wv, Qw, Kw, VTw, qScale);

  flash_attn_kernel<<<dim3(cS / 128, cH, cB), 256, 0, stream>>>(Qw, Kw, VTw, AOw);

  out_proj_kernel<<<dim3(cM / 128, cE / 64), 256, 0, stream>>>(AOw, Wo, bo, out);
}
